// Encoder_80307298501354
// MI455X (gfx1250) — compile-verified, matrix operand order checked by reading, not run on hardware
//
#include <hip/hip_runtime.h>
#include <hip/hip_bf16.h>

// ---------------------------------------------------------------------------
// Transformer encoder (B=4, S=512, D=1024, F=4096, H=16, depth=64, L=6) for
// MI455X (gfx1250, wave32). All GEMMs run on v_wmma_f32_16x16x32_bf16 with
// fp32 accumulate. Tiles are staged to LDS with global_load_async_to_lds_b128
// (ASYNCcnt) where no conversion is needed; fp32 weights are converted to
// packed bf16 pairs during staging. Workspace requirement: ~168 MB.
// ---------------------------------------------------------------------------

typedef __attribute__((ext_vector_type(16))) __bf16 v16bf;
typedef __attribute__((ext_vector_type(8)))  float  v8f;

__device__ __forceinline__ unsigned short f2bf(float f) {
  unsigned u = __float_as_uint(f);
  u += 0x7FFFu + ((u >> 16) & 1u);          // round-to-nearest-even
  return (unsigned short)(u >> 16);
}

// pack two f32 -> packed bf16 (lo in [15:0], hi in [31:16])
__device__ __forceinline__ unsigned pack2_bf16(float lo, float hi) {
#if __has_builtin(__builtin_amdgcn_cvt_pk_bf16_f32)
  typedef __attribute__((ext_vector_type(2))) __bf16 v2bf;
  v2bf r = __builtin_amdgcn_cvt_pk_bf16_f32(lo, hi);
  union { v2bf v; unsigned u; } c; c.v = r; return c.u;
#else
  return ((unsigned)f2bf(hi) << 16) | (unsigned)f2bf(lo);
#endif
}

// ---------------------------------------------------------------------------
// Generic tiled WMMA GEMM (BK = 64 -> 2 wmma K-steps per stage).
//   A: bf16 [M,K] row-major, optionally batched; staged via async-to-LDS.
//   B: BSRC=0 -> fp32 [K,N] row-major (weights; convert to packed bf16)
//      BSRC=1 -> bf16 [K,N] row-major (batched; pair-packed transpose)
//      BSRC=2 -> bf16 [N,K] row-major (NT; staged via async-to-LDS)
//   EPI: 0 = fp32 out [M,N], +bias
//        1 = bf16 out scattered to [B,H,S,64] head layout, +bias   (QKV)
//        2 = bf16 out [M,N], +bias, ReLU                           (FFN1)
//        3 = bf16 out merged heads [B,S,1024], batch z=(b*16+h)    (P@V)
//        4 = fp32 out logits [z][M][N] = acc*0.125 + mask*(-1e9)   (Q@K^T)
// Assumes M%BM==0, N%BN==0, K%64==0 (true for every call below).
// ---------------------------------------------------------------------------
template<int BM, int BN, int WAVES_M, int WAVES_N, int BSRC, int EPI>
__global__ __launch_bounds__(256)
void gemm_wmma(const unsigned short* __restrict__ A,
               const void* __restrict__ Bsrc,
               const float* __restrict__ bias,
               void* __restrict__ Cout,
               const float* __restrict__ mask,
               int M, int N, int K,
               long long aBatch, long long bBatch)
{
  constexpr int BK  = 64;
  constexpr int LDA = BK + 8;                 // 144B rows: banked, 16B-aligned
  constexpr int WTM = BM / WAVES_M;
  constexpr int WTN = BN / WAVES_N;
  constexpr int TM  = WTM / 16;
  constexpr int TN  = WTN / 16;

  __shared__ unsigned short As[BM * LDA];     // A tile  [m][k]
  __shared__ unsigned short Bt[BN * LDA];     // B tile transposed: [n][k]

  const int tid  = threadIdx.x;
  const int lane = tid & 31;
  const int wid  = tid >> 5;
  const int wm   = wid / WAVES_N;
  const int wn   = wid % WAVES_N;

  const int bm0 = blockIdx.y * BM;
  const int bn0 = blockIdx.x * BN;
  const int z   = blockIdx.z;

  const unsigned short* Ab = A + (long long)z * aBatch;

  v8f acc[TM][TN];
#pragma unroll
  for (int i = 0; i < TM; i++)
#pragma unroll
    for (int j = 0; j < TN; j++)
      acc[i][j] = (v8f){0.f, 0.f, 0.f, 0.f, 0.f, 0.f, 0.f, 0.f};

  for (int k0 = 0; k0 < K; k0 += BK) {
    // ---- stage A tile: BM x 64 bf16, async memory->LDS (16B chunks) -----
    {
      constexpr int CH = BM * (BK / 8);
      for (int c = tid; c < CH; c += 256) {
        const int r  = c >> 3;
        const int kc = (c & 7) * 8;
        const unsigned lofs = (unsigned)(unsigned long long)&As[r * LDA + kc];
        const unsigned long long ga =
            (unsigned long long)(Ab + (long long)(bm0 + r) * K + k0 + kc);
        asm volatile("global_load_async_to_lds_b128 %0, %1, off"
                     :: "v"(lofs), "v"(ga) : "memory");
      }
    }
    // ---- stage B tile into transposed LDS layout ------------------------
    if constexpr (BSRC == 0) {                // fp32 [K,N] weights -> bf16
      const float* Bf = (const float*)Bsrc;
      constexpr int CH = (BK / 2) * (BN / 4); // chunk = (k-pair, 4 cols)
      for (int c = tid; c < CH; c += 256) {
        const int kp = c / (BN / 4);          // k-pair index, k = 2*kp
        const int nc = (c % (BN / 4)) * 4;
        const float* g0 = Bf + (long long)(k0 + 2 * kp) * N + bn0 + nc;
        if (k0 + BK < K) __builtin_prefetch(g0 + (long long)BK * N, 0, 1);
        float4 da = *(const float4*)g0;
        float4 db = *(const float4*)(g0 + N);
        *(unsigned*)&Bt[(nc + 0) * LDA + 2 * kp] = pack2_bf16(da.x, db.x);
        *(unsigned*)&Bt[(nc + 1) * LDA + 2 * kp] = pack2_bf16(da.y, db.y);
        *(unsigned*)&Bt[(nc + 2) * LDA + 2 * kp] = pack2_bf16(da.z, db.z);
        *(unsigned*)&Bt[(nc + 3) * LDA + 2 * kp] = pack2_bf16(da.w, db.w);
      }
    } else if constexpr (BSRC == 1) {         // bf16 [K,N] -> pair-packed
      const unsigned short* Bh = (const unsigned short*)Bsrc + (long long)z * bBatch;
      constexpr int CH = (BK / 2) * (BN / 8); // chunk = (k-pair, 8 cols)
      for (int c = tid; c < CH; c += 256) {
        const int kp = c / (BN / 8);
        const int nc = (c % (BN / 8)) * 8;
        uint4 da = *(const uint4*)(Bh + (long long)(k0 + 2 * kp) * N + bn0 + nc);
        uint4 db = *(const uint4*)(Bh + (long long)(k0 + 2 * kp + 1) * N + bn0 + nc);
        const unsigned short* pa = (const unsigned short*)&da;
        const unsigned short* pb = (const unsigned short*)&db;
#pragma unroll
        for (int j = 0; j < 8; j++)
          *(unsigned*)&Bt[(nc + j) * LDA + 2 * kp] =
              ((unsigned)pb[j] << 16) | (unsigned)pa[j];
      }
    } else {                                  // BSRC==2: bf16 [N,K] (NT), async
      const unsigned short* Bh = (const unsigned short*)Bsrc + (long long)z * bBatch;
      constexpr int CH = BN * (BK / 8);
      for (int c = tid; c < CH; c += 256) {
        const int r  = c >> 3;
        const int kc = (c & 7) * 8;
        const unsigned lofs = (unsigned)(unsigned long long)&Bt[r * LDA + kc];
        const unsigned long long ga =
            (unsigned long long)(Bh + (long long)(bn0 + r) * K + k0 + kc);
        asm volatile("global_load_async_to_lds_b128 %0, %1, off"
                     :: "v"(lofs), "v"(ga) : "memory");
      }
    }
    asm volatile("s_wait_asynccnt 0x0" ::: "memory");
    __syncthreads();

    // ---- fragments + WMMA: two K=32 sub-steps per stage -----------------
    // 16-bit A layout (ISA 7.12.2): lane<16 -> row=lane, K in [0,8)+[16,24);
    // lane>=16 -> row=lane-16, K in [8,16)+[24,32). B identical on the
    // transposed LDS tile.
    const int fr = lane & 15;
    const int kh = (lane >> 4) * 8;
#pragma unroll
    for (int ks = 0; ks < BK; ks += 32) {
      v16bf afr[TM], bfr[TN];
#pragma unroll
      for (int i = 0; i < TM; i++) {
        const unsigned short* p = &As[(wm * WTM + i * 16 + fr) * LDA + ks + kh];
        union { uint4 u[2]; v16bf v; } u;
        u.u[0] = *(const uint4*)p;
        u.u[1] = *(const uint4*)(p + 16);
        afr[i] = u.v;
      }
#pragma unroll
      for (int j = 0; j < TN; j++) {
        const unsigned short* p = &Bt[(wn * WTN + j * 16 + fr) * LDA + ks + kh];
        union { uint4 u[2]; v16bf v; } u;
        u.u[0] = *(const uint4*)p;
        u.u[1] = *(const uint4*)(p + 16);
        bfr[j] = u.v;
      }
#pragma unroll
      for (int i = 0; i < TM; i++)
#pragma unroll
        for (int j = 0; j < TN; j++)
          acc[i][j] = __builtin_amdgcn_wmma_f32_16x16x32_bf16(
              false, afr[i], false, bfr[j], (short)0, acc[i][j], false, false);
    }
    __syncthreads();
  }

  // ---- epilogue ---------------------------------------------------------
  // C layout: VGPR e of lane l holds (row = 8*(l/16)+e, col = l%16).
  const int er = (lane >> 4) * 8;
  const int ec = lane & 15;
#pragma unroll
  for (int i = 0; i < TM; i++) {
#pragma unroll
    for (int j = 0; j < TN; j++) {
      const int gcol = bn0 + wn * WTN + j * 16 + ec;
#pragma unroll
      for (int e = 0; e < 8; e++) {
        const int grow = bm0 + wm * WTM + i * 16 + er + e;
        float v = acc[i][j][e];
        if constexpr (EPI == 0) {
          v += bias[gcol];
          ((float*)Cout)[(long long)grow * N + gcol] = v;
        } else if constexpr (EPI == 1) {       // QKV -> [B,H,S,64] bf16
          v += bias[gcol];
          const int b = grow >> 9, s = grow & 511;
          const int h = gcol >> 6, dd = gcol & 63;
          ((unsigned short*)Cout)[((((long long)b * 16 + h) * 512 + s) << 6) + dd] = f2bf(v);
        } else if constexpr (EPI == 2) {       // FFN1: ReLU, bf16
          v += bias[gcol];
          v = v > 0.f ? v : 0.f;
          ((unsigned short*)Cout)[(long long)grow * N + gcol] = f2bf(v);
        } else if constexpr (EPI == 3) {       // P@V: merge heads -> [B,S,1024]
          const int b = z >> 4, h = z & 15;
          ((unsigned short*)Cout)[((long long)(b * 512 + grow)) * 1024 + h * 64 + gcol] = f2bf(v);
        } else {                               // logits: scale + mask
          const int b = z >> 4;
          v = v * 0.125f + mask[(long long)b * 512 + gcol] * (-1e9f);
          ((float*)Cout)[(long long)z * 512 * 512 + (long long)grow * 512 + gcol] = v;
        }
      }
    }
  }
}

// ---------------------------------------------------------------------------
// Row softmax over 512 columns: one wave32 per row, fp32 in, bf16 probs out.
// ---------------------------------------------------------------------------
__global__ __launch_bounds__(256)
void softmax_rows(const float* __restrict__ logits, unsigned short* __restrict__ probs)
{
  const int row  = blockIdx.x * 8 + (threadIdx.x >> 5);
  const int lane = threadIdx.x & 31;
  const float* lp = logits + (long long)row * 512;
  float v[16];
  float m = -3.4e38f;
#pragma unroll
  for (int i = 0; i < 16; i++) { v[i] = lp[lane + i * 32]; m = fmaxf(m, v[i]); }
#pragma unroll
  for (int o = 16; o > 0; o >>= 1) m = fmaxf(m, __shfl_xor(m, o, 32));
  float s = 0.f;
#pragma unroll
  for (int i = 0; i < 16; i++) { v[i] = __expf(v[i] - m); s += v[i]; }
#pragma unroll
  for (int o = 16; o > 0; o >>= 1) s += __shfl_xor(s, o, 32);
  const float inv = 1.0f / s;
  unsigned short* pp = probs + (long long)row * 512;
#pragma unroll
  for (int i = 0; i < 8; i++)
    *(unsigned*)&pp[lane * 2 + i * 64] = pack2_bf16(v[2 * i] * inv, v[2 * i + 1] * inv);
}

// ---------------------------------------------------------------------------
// Fused residual-add + LayerNorm over D=1024; one block per row; emits fp32
// (next residual) and bf16 (next GEMM A operand).
// ---------------------------------------------------------------------------
__global__ __launch_bounds__(256)
void layernorm_add(const float* __restrict__ resid, const float* __restrict__ delta,
                   const float* __restrict__ gamma, const float* __restrict__ beta,
                   float* __restrict__ outf, unsigned short* __restrict__ outh)
{
  __shared__ float redA[8], redB[8];
  const int row = blockIdx.x;
  const int tid = threadIdx.x, lane = tid & 31, wid = tid >> 5;
  const float* rp = resid + (long long)row * 1024;
  const float* dp = delta + (long long)row * 1024;
  float x[4];
  float s = 0.f;
#pragma unroll
  for (int i = 0; i < 4; i++) { const int c = tid * 4 + i; x[i] = rp[c] + dp[c]; s += x[i]; }
#pragma unroll
  for (int o = 16; o > 0; o >>= 1) s += __shfl_xor(s, o, 32);
  if (lane == 0) redA[wid] = s;
  __syncthreads();
  if (tid == 0) { float t = 0.f; for (int i = 0; i < 8; i++) t += redA[i]; redA[0] = t; }
  __syncthreads();
  const float mean = redA[0] * (1.0f / 1024.0f);
  float vs = 0.f;
#pragma unroll
  for (int i = 0; i < 4; i++) { const float d = x[i] - mean; vs += d * d; }
#pragma unroll
  for (int o = 16; o > 0; o >>= 1) vs += __shfl_xor(vs, o, 32);
  if (lane == 0) redB[wid] = vs;
  __syncthreads();
  if (tid == 0) { float t = 0.f; for (int i = 0; i < 8; i++) t += redB[i]; redB[0] = t; }
  __syncthreads();
  const float rstd = rsqrtf(redB[0] * (1.0f / 1024.0f) + 1e-6f);
  float o4[4];
#pragma unroll
  for (int i = 0; i < 4; i++) {
    const int c = tid * 4 + i;
    o4[i] = (x[i] - mean) * rstd * gamma[c] + beta[c];
    outf[(long long)row * 1024 + c] = o4[i];
  }
  unsigned* oh = (unsigned*)(outh + (long long)row * 1024 + tid * 4);
  oh[0] = pack2_bf16(o4[0], o4[1]);
  oh[1] = pack2_bf16(o4[2], o4[3]);
}

// ---------------------------------------------------------------------------
// x + sinusoidal positional encoding; emits fp32 + bf16 mirror.
// ---------------------------------------------------------------------------
__global__ __launch_bounds__(256)
void pos_encode(const float* __restrict__ x, float* __restrict__ hf,
                unsigned short* __restrict__ hb)
{
  const long long idx = (long long)blockIdx.x * 256 + threadIdx.x;  // B*S*D
  const int d = (int)(idx & 1023);
  const int s = (int)((idx >> 10) & 511);
  const float expo = (float)(2 * (d >> 1)) * (1.0f / 1024.0f);
  const float freq = __powf(10000.0f, -expo);
  const float a = (float)s * freq;
  const float pe = (d & 1) ? __cosf(a) : __sinf(a);
  const float v = x[idx] + pe;
  hf[idx] = v;
  hb[idx] = f2bf(v);
}

// ---------------------------------------------------------------------------
extern "C" void kernel_launch(void* const* d_in, const int* in_sizes, int n_in,
                              void* d_out, int out_size, void* d_ws, size_t ws_size,
                              hipStream_t stream)
{
  const float* x    = (const float*)d_in[0];
  const float* mask = (const float*)d_in[1];
  const float* Wq   = (const float*)d_in[2];
  const float* bq   = (const float*)d_in[3];
  const float* Wk   = (const float*)d_in[4];
  const float* bk   = (const float*)d_in[5];
  const float* Wv   = (const float*)d_in[6];
  const float* bv   = (const float*)d_in[7];
  const float* Wo   = (const float*)d_in[8];
  const float* bo   = (const float*)d_in[9];
  const float* W1   = (const float*)d_in[10];
  const float* b1   = (const float*)d_in[11];
  const float* W2   = (const float*)d_in[12];
  const float* b2   = (const float*)d_in[13];
  const float* g1   = (const float*)d_in[14];
  const float* be1  = (const float*)d_in[15];
  const float* g2   = (const float*)d_in[16];
  const float* be2  = (const float*)d_in[17];

  const int B = 4, S = 512, D = 1024, F = 4096, H = 16, L = 6;
  const int M = B * S;                         // 2048

  // bump allocator over workspace (~168 MB total)
  char*  ws  = (char*)d_ws;
  size_t off = 0;
  auto alloc = [&](size_t bytes) -> char* {
    char* p = ws + off;
    off += (bytes + 255) & ~(size_t)255;
    return p;
  };
  float*          hF     = (float*)alloc((size_t)M * D * 4);
  unsigned short* hB     = (unsigned short*)alloc((size_t)M * D * 2);
  float*          h1F    = (float*)alloc((size_t)M * D * 4);
  unsigned short* h1B    = (unsigned short*)alloc((size_t)M * D * 2);
  float*          d1F    = (float*)alloc((size_t)M * D * 4);   // o-proj out
  float*          d2F    = (float*)alloc((size_t)M * D * 4);   // ffn2 out
  unsigned short* qB     = (unsigned short*)alloc((size_t)M * D * 2);
  unsigned short* kB     = (unsigned short*)alloc((size_t)M * D * 2);
  unsigned short* vB     = (unsigned short*)alloc((size_t)M * D * 2);
  float*          logits = (float*)alloc((size_t)B * H * S * S * 4);   // 64 MB
  unsigned short* probs  = (unsigned short*)alloc((size_t)B * H * S * S * 2);
  unsigned short* attnB  = (unsigned short*)alloc((size_t)M * D * 2);
  unsigned short* f1B    = (unsigned short*)alloc((size_t)M * F * 2);
  (void)ws_size; (void)in_sizes; (void)n_in; (void)out_size;

  const dim3 blk(256);
  pos_encode<<<(M * D) / 256, blk, 0, stream>>>(x, hF, hB);

  for (int l = 0; l < L; l++) {
    const float* wq = Wq + (size_t)l * D * D;  const float* bql = bq + (size_t)l * D;
    const float* wk = Wk + (size_t)l * D * D;  const float* bkl = bk + (size_t)l * D;
    const float* wv = Wv + (size_t)l * D * D;  const float* bvl = bv + (size_t)l * D;
    const float* wo = Wo + (size_t)l * D * D;  const float* bol = bo + (size_t)l * D;
    const float* w1 = W1 + (size_t)l * D * F;  const float* b1l = b1 + (size_t)l * F;
    const float* w2 = W2 + (size_t)l * F * D;  const float* b2l = b2 + (size_t)l * D;
    const float* g1l = g1 + (size_t)l * D;     const float* be1l = be1 + (size_t)l * D;
    const float* g2l = g2 + (size_t)l * D;     const float* be2l = be2 + (size_t)l * D;

    // Q, K, V projections -> head-major bf16
    gemm_wmma<128,128,2,4,0,1><<<dim3(D/128, M/128, 1), blk, 0, stream>>>(
        hB, wq, bql, qB, nullptr, M, D, D, 0, 0);
    gemm_wmma<128,128,2,4,0,1><<<dim3(D/128, M/128, 1), blk, 0, stream>>>(
        hB, wk, bkl, kB, nullptr, M, D, D, 0, 0);
    gemm_wmma<128,128,2,4,0,1><<<dim3(D/128, M/128, 1), blk, 0, stream>>>(
        hB, wv, bvl, vB, nullptr, M, D, D, 0, 0);

    // logits = Q @ K^T * scale + mask*(-1e9), batched over B*H
    gemm_wmma<128,128,2,4,2,4><<<dim3(S/128, S/128, B*H), blk, 0, stream>>>(
        qB, kB, nullptr, logits, mask, S, S, 64,
        (long long)S * 64, (long long)S * 64);
    softmax_rows<<<(B * H * S) / 8, blk, 0, stream>>>(logits, probs);

    // O = P @ V, merge heads
    gemm_wmma<128,64,4,2,1,3><<<dim3(1, S/128, B*H), blk, 0, stream>>>(
        probs, vB, nullptr, attnB, nullptr, S, 64, S,
        (long long)S * S, (long long)S * 64);

    // output projection + LN1
    gemm_wmma<128,128,2,4,0,0><<<dim3(D/128, M/128, 1), blk, 0, stream>>>(
        attnB, wo, bol, d1F, nullptr, M, D, D, 0, 0);
    layernorm_add<<<M, blk, 0, stream>>>(hF, d1F, g1l, be1l, h1F, h1B);

    // FFN
    gemm_wmma<128,128,2,4,0,2><<<dim3(F/128, M/128, 1), blk, 0, stream>>>(
        h1B, w1, b1l, f1B, nullptr, M, F, D, 0, 0);
    gemm_wmma<128,128,2,4,0,0><<<dim3(D/128, M/128, 1), blk, 0, stream>>>(
        f1B, w2, b2l, d2F, nullptr, M, D, F, 0, 0);

    float* outTarget = (l == L - 1) ? (float*)d_out : hF;
    layernorm_add<<<M, blk, 0, stream>>>(h1F, d2F, g2l, be2l, outTarget, hB);
  }
}
